// AttentionGNNEncoder_25331717112515
// MI455X (gfx1250) — matrix operand order repaired and checked by reading, not compiled, not run on hardware
//
#include <hip/hip_runtime.h>

// ---------------------------------------------------------------------------
// GAT encoder for gfx1250 (MI455X).
// GEMMs run on v_wmma_f32_16x16x32_bf16 with a bf16(hi)+bf16(lo) split of both
// operands (3 WMMAs per tile) for ~fp32 accuracy at matrix-core rate.
// Edge softmax + aggregation use f32 global atomics; the 102MB node-feature
// working set is L2-resident (192MB L2), so scatters are L2-bound not HBM.
// ---------------------------------------------------------------------------

typedef __attribute__((ext_vector_type(16))) __bf16 v16bf;
typedef __attribute__((ext_vector_type(8)))  float  v8f;

static constexpr int       Nn      = 50000;
static constexpr long long Ee      = 800000;
static constexpr int       CH      = 128;   // HID
static constexpr int       NGRAPH  = 50;

// ---------------------------------------------------------------- utilities
__global__ void fill_f32(float* __restrict__ p, float v, long long n) {
  long long i = (long long)blockIdx.x * blockDim.x + threadIdx.x;
  if (i < n) p[i] = v;
}

__device__ __forceinline__ void atomicMaxF(float* addr, float val) {
  if (val >= 0.0f) atomicMax((int*)addr, __float_as_int(val));
  else             atomicMin((unsigned int*)addr, (unsigned int)__float_as_int(val));
}

// ------------------------------------------------- weight pack: f32 -> bf16x2
// Destination layout matches the WMMA B-operand (32x16 bf16) lane layout:
// tile index = ktile*(M/16)+mtile, within tile: [lane][i] where
//   K = ktile*32 + (i>=8 ? 16 : 0) + (lane>=16 ? 8 : 0) + (i&7),  col = mtile*16 + (lane&15)
__global__ void pack_w(const float* __restrict__ W, __bf16* __restrict__ Wp,
                       int K, int M) {
  int tid = blockIdx.x * blockDim.x + threadIdx.x;
  int total = K * M;
  if (tid >= total) return;
  int i    = tid & 15;
  int lane = (tid >> 4) & 31;
  int rest = tid >> 9;
  int mtiles = M >> 4;
  int mtile = rest % mtiles;
  int ktile = rest / mtiles;
  int kk = ktile * 32 + (i >> 3) * 16 + (lane >> 4) * 8 + (i & 7);
  int mm = mtile * 16 + (lane & 15);
  float w  = W[(long long)kk * M + mm];
  __bf16 h = (__bf16)w;
  Wp[tid]          = h;                         // hi part
  Wp[total + tid]  = (__bf16)(w - (float)h);    // lo residual
}

// --------------------------------------------------------- WMMA GEMM (bf16x3)
// C[N x M] = A[N x K] @ W[K x M]; one wave per 16x128 tile.
__global__ void __launch_bounds__(32)
gemm_wmma_bf16x3(const float* __restrict__ A, const __bf16* __restrict__ Bhi,
                 const __bf16* __restrict__ Blo, float* __restrict__ C,
                 int K, int M) {
  const int lane = threadIdx.x;
  const int row  = lane & 15;
  const int hi   = lane >> 4;
  const long long rbase = (long long)blockIdx.x * 16;
  const int cbase  = blockIdx.y * 128;
  const int mtiles = M >> 4;

  v8f acc[8] = {};

  const float* arow = A + (rbase + row) * (long long)K;

  for (int k = 0; k < K; k += 32) {
    // A fragment (16x32 bf16 layout): lanes 0-15 hold K {0..7,16..23},
    // lanes 16-31 hold K {8..15,24..31}
    const float* ap = arow + k + hi * 8;
    float4 p0 = *(const float4*)(ap);
    float4 p1 = *(const float4*)(ap + 4);
    float4 p2 = *(const float4*)(ap + 16);
    float4 p3 = *(const float4*)(ap + 20);
    float av[16] = {p0.x, p0.y, p0.z, p0.w, p1.x, p1.y, p1.z, p1.w,
                    p2.x, p2.y, p2.z, p2.w, p3.x, p3.y, p3.z, p3.w};
    v16bf ah, al;
#pragma unroll
    for (int i = 0; i < 16; ++i) {
      __bf16 h = (__bf16)av[i];
      ah[i] = h;
      al[i] = (__bf16)(av[i] - (float)h);
    }

    const long long boff =
        ((long long)(k >> 5) * mtiles + (cbase >> 4)) * 512 + lane * 16;
#pragma unroll
    for (int t = 0; t < 8; ++t) {
      v16bf bh = *(const v16bf*)(Bhi + boff + t * 512);
      v16bf bl = *(const v16bf*)(Blo + boff + t * 512);
      acc[t] = __builtin_amdgcn_wmma_f32_16x16x32_bf16(
          false, ah, false, bh, (short)0, acc[t], false, false);
      acc[t] = __builtin_amdgcn_wmma_f32_16x16x32_bf16(
          false, al, false, bh, (short)0, acc[t], false, false);
      acc[t] = __builtin_amdgcn_wmma_f32_16x16x32_bf16(
          false, ah, false, bl, (short)0, acc[t], false, false);
    }
  }

  // C/D layout: element j of lane -> (m = 8*hi + j, n = lane&15)
#pragma unroll
  for (int t = 0; t < 8; ++t) {
    float* cp = C + (rbase + hi * 8) * (long long)M + cbase + t * 16 + row;
#pragma unroll
    for (int j = 0; j < 8; ++j) cp[(long long)j * M] = acc[t][j];
  }
}

// ----------------------------------------------- attention projections (a.h)
// one wave per (node, head); ch = 128 fixed
__global__ void attn_alpha(const float* __restrict__ h, const float* __restrict__ as,
                           const float* __restrict__ ad, float* __restrict__ als,
                           float* __restrict__ ald, int heads, int ntasks) {
  int wid  = (int)((blockIdx.x * (long long)blockDim.x + threadIdx.x) >> 5);
  int lane = threadIdx.x & 31;
  if (wid >= ntasks) return;
  int node = wid / heads, hd = wid - node * heads;
  const float* hp  = h + ((long long)node * heads + hd) * CH;
  const float* asp = as + hd * CH;
  const float* adp = ad + hd * CH;
  float s0 = 0.f, s1 = 0.f;
#pragma unroll
  for (int c = lane; c < CH; c += 32) {
    float v = hp[c];
    s0 += v * asp[c];
    s1 += v * adp[c];
  }
  for (int off = 16; off; off >>= 1) {
    s0 += __shfl_xor(s0, off, 32);
    s1 += __shfl_xor(s1, off, 32);
  }
  if (lane == 0) { als[wid] = s0; ald[wid] = s1; }
}

// --------------------------------------------- edge logits + segment max(dst)
__global__ void edge_logits_max(const int* __restrict__ srcI, const int* __restrict__ dstI,
                                const float* __restrict__ als, const float* __restrict__ ald,
                                float* __restrict__ logit, float* __restrict__ emax,
                                int heads, int total) {
  int tid = blockIdx.x * blockDim.x + threadIdx.x;
  if (tid >= total) return;
  int e = tid / heads, hd = tid - e * heads;
  int s = srcI[e], d = dstI[e];
  float v = als[s * heads + hd] + ald[d * heads + hd];
  v = v > 0.f ? v : 0.2f * v;                 // leaky_relu(0.2)
  logit[tid] = v;
  atomicMaxF(&emax[d * heads + hd], v);
}

// ------------------------------------------------ exp + segment sum of denom
__global__ void edge_exp_denom(const int* __restrict__ dstI, float* __restrict__ logit,
                               const float* __restrict__ emax, float* __restrict__ denom,
                               int heads, int total) {
  int tid = blockIdx.x * blockDim.x + threadIdx.x;
  if (tid >= total) return;
  int e = tid / heads, hd = tid - e * heads;
  int d = dstI[e];
  float ex = expf(logit[tid] - emax[d * heads + hd]);
  logit[tid] = ex;                             // in place: logit -> exp value
  atomicAdd(&denom[d * heads + hd], ex);
}

// ---------------------------------- message scatter: out[dst] += h[src]*alpha
// 4 channels per thread (float4 gather + 4 f32 atomics), L2-resident
__global__ void edge_scatter(const int* __restrict__ srcI, const int* __restrict__ dstI,
                             const float* __restrict__ h, const float* __restrict__ ex,
                             const float* __restrict__ denom, float* __restrict__ out,
                             int heads, long long total) {
  long long tid = (long long)blockIdx.x * blockDim.x + threadIdx.x;
  if (tid >= total) return;
  int perEdge = heads * 32;                    // heads*128/4
  long long e = tid / perEdge;
  int j  = (int)(tid - e * perEdge);
  int hd = j >> 5;
  int c4 = (j & 31) << 2;
  int s = srcI[e], d = dstI[e];
  float a = ex[e * heads + hd] / (denom[(long long)d * heads + hd] + 1e-16f);
  float4 hv = *(const float4*)(h + ((long long)s * heads + hd) * CH + c4);
  float* op = out + ((long long)d * heads + hd) * CH + c4;
  atomicAdd(op + 0, hv.x * a);
  atomicAdd(op + 1, hv.y * a);
  atomicAdd(op + 2, hv.z * a);
  atomicAdd(op + 3, hv.w * a);
}

// ------------------------------------------------------------- bias (+ ELU)
__global__ void bias_act(float* __restrict__ o, const float* __restrict__ b,
                         int fdim, long long total, int elu) {
  long long tid = (long long)blockIdx.x * blockDim.x + threadIdx.x;
  if (tid >= total) return;
  float v = o[tid] + b[(int)(tid % fdim)];
  if (elu) v = v > 0.f ? v : expm1f(v);
  o[tid] = v;
}

// -------------------------------------------------------------- mean pooling
__global__ void pool_cnt(const int* __restrict__ bt, float* __restrict__ gcnt, int n) {
  int i = blockIdx.x * blockDim.x + threadIdx.x;
  if (i < n) atomicAdd(&gcnt[bt[i]], 1.0f);
}
__global__ void pool_sum(const float* __restrict__ ne, const int* __restrict__ bt,
                         float* __restrict__ gsum, long long total) {
  long long tid = (long long)blockIdx.x * blockDim.x + threadIdx.x;
  if (tid >= total) return;
  int node = (int)(tid >> 7), c = (int)(tid & 127);
  atomicAdd(&gsum[(long long)bt[node] * CH + c], ne[tid]);
}
__global__ void pool_div(const float* __restrict__ gsum, const float* __restrict__ gcnt,
                         float* __restrict__ gemb, int total) {
  int tid = blockIdx.x * blockDim.x + threadIdx.x;
  if (tid >= total) return;
  gemb[tid] = gsum[tid] / fmaxf(gcnt[tid >> 7], 1.0f);
}

// ------------------------------------------------------------------- driver
static inline unsigned nblk(long long n) { return (unsigned)((n + 255) / 256); }

static void run_layer(const float* feat, const float* W, const float* avs,
                      const float* avd, const float* bias, float* h, float* out,
                      int K, int M, int heads, int elu, const int* ei, __bf16* wp,
                      float* als, float* ald, float* emax, float* denom, float* exb,
                      hipStream_t stream) {
  long long KM = (long long)K * M;
  pack_w<<<nblk(KM), 256, 0, stream>>>(W, wp, K, M);
  dim3 g(Nn / 16, M / 128);
  gemm_wmma_bf16x3<<<g, 32, 0, stream>>>(feat, wp, wp + KM, h, K, M);

  long long tasks = (long long)Nn * heads;
  attn_alpha<<<nblk(tasks * 32), 256, 0, stream>>>(h, avs, avd, als, ald, heads,
                                                   (int)tasks);
  fill_f32<<<nblk(tasks), 256, 0, stream>>>(emax, -__builtin_huge_valf(), tasks);
  fill_f32<<<nblk(tasks), 256, 0, stream>>>(denom, 0.0f, tasks);
  fill_f32<<<nblk((long long)Nn * M), 256, 0, stream>>>(out, 0.0f, (long long)Nn * M);

  long long eh = Ee * heads;
  edge_logits_max<<<nblk(eh), 256, 0, stream>>>(ei, ei + Ee, als, ald, exb, emax,
                                                heads, (int)eh);
  edge_exp_denom<<<nblk(eh), 256, 0, stream>>>(ei + Ee, exb, emax, denom, heads,
                                               (int)eh);
  long long tot = Ee * heads * 32;   // (heads*128)/4 per edge
  edge_scatter<<<nblk(tot), 256, 0, stream>>>(ei, ei + Ee, h, exb, denom, out,
                                              heads, tot);
  bias_act<<<nblk((long long)Nn * M), 256, 0, stream>>>(out, bias, M,
                                                        (long long)Nn * M, elu);
}

extern "C" void kernel_launch(void* const* d_in, const int* in_sizes, int n_in,
                              void* d_out, int out_size, void* d_ws, size_t ws_size,
                              hipStream_t stream) {
  const float* x   = (const float*)d_in[0];
  const int*   ei  = (const int*)d_in[1];     // [2,E] int32 (jax x64 disabled)
  const int*   bt  = (const int*)d_in[2];
  const float* W1  = (const float*)d_in[3];
  const float* a1s = (const float*)d_in[4];
  const float* a1d = (const float*)d_in[5];
  const float* b1  = (const float*)d_in[6];
  const float* W2  = (const float*)d_in[7];
  const float* a2s = (const float*)d_in[8];
  const float* a2d = (const float*)d_in[9];
  const float* b2  = (const float*)d_in[10];
  const float* W3  = (const float*)d_in[11];
  const float* a3s = (const float*)d_in[12];
  const float* a3d = (const float*)d_in[13];
  const float* b3  = (const float*)d_in[14];

  float* ws    = (float*)d_ws;
  float* hA    = ws;                              // N*512
  float* hB    = hA   + (long long)Nn * 512;      // N*512
  float* als   = hB   + (long long)Nn * 512;      // N*4
  float* ald   = als  + (long long)Nn * 4;
  float* emax  = ald  + (long long)Nn * 4;
  float* denom = emax + (long long)Nn * 4;
  float* exb   = denom + (long long)Nn * 4;       // E*4
  float* gsum  = exb  + Ee * 4;                   // 50*128
  float* gcnt  = gsum + NGRAPH * CH;              // 50 (+pad)
  __bf16* wp   = (__bf16*)(gcnt + 64);            // 2 * 512*512 bf16 (hi|lo)

  float* node_emb  = (float*)d_out;                         // N*128
  float* graph_emb = node_emb + (long long)Nn * CH;         // 50*128

  // Layer 1: [N,256] -> [N,512], 4 heads, ELU
  run_layer(x, W1, a1s, a1d, b1, hA, hB, 256, 512, 4, 1, ei, wp,
            als, ald, emax, denom, exb, stream);
  // Layer 2: [N,512] -> [N,512], 4 heads, ELU  (h in hA, out reuses hB)
  run_layer(hB, W2, a2s, a2d, b2, hA, hB, 512, 512, 4, 1, ei, wp,
            als, ald, emax, denom, exb, stream);
  // Layer 3: [N,512] -> [N,128], 1 head, no ELU, result straight into d_out
  run_layer(hB, W3, a3s, a3d, b3, hA, node_emb, 512, 128, 1, 0, ei, wp,
            als, ald, emax, denom, exb, stream);

  // Global mean pool per graph
  fill_f32<<<nblk(NGRAPH * CH), 256, 0, stream>>>(gsum, 0.0f, NGRAPH * CH);
  fill_f32<<<nblk(NGRAPH), 256, 0, stream>>>(gcnt, 0.0f, NGRAPH);
  pool_cnt<<<nblk(Nn), 256, 0, stream>>>(bt, gcnt, Nn);
  pool_sum<<<nblk((long long)Nn * CH), 256, 0, stream>>>(node_emb, bt, gsum,
                                                         (long long)Nn * CH);
  pool_div<<<nblk(NGRAPH * CH), 256, 0, stream>>>(gsum, gcnt, graph_emb,
                                                  NGRAPH * CH);
}
